// SAGEEncoder_54571854463793
// MI455X (gfx1250) — compile-verified, matrix operand order checked by reading, not run on hardware
//
#include <hip/hip_runtime.h>
#include <hip/hip_bf16.h>
#include <stdint.h>
#include <stddef.h>

#define N_NODES 100000
#define D 128

typedef __attribute__((ext_vector_type(16))) __bf16 v16bf;
typedef __attribute__((ext_vector_type(8)))  float  v8f;

// ---------------------------------------------------------------------------
// Workspace layout (bytes); requires ws_size >= ~103.1 MB
// ---------------------------------------------------------------------------
#define OFF_AGG   ((size_t)0)
#define SZ_AGG    ((size_t)N_NODES * D * 4)            // 51,200,000
#define OFF_DEG   (OFF_AGG + SZ_AGG)                   // 51,200,000 (256B aligned)
#define SZ_DEG    ((size_t)409600)                     // N*4 rounded up
#define OFF_H     (OFF_DEG + SZ_DEG)                   // 51,609,600
#define SZ_H      ((size_t)N_NODES * D * 4)
#define OFF_WF    (OFF_H + SZ_H)                       // 102,809,600 (256B aligned)
#define WF_ELEMS  ((size_t)16384)                      // 8 nt * 4 kk * 32 lanes * 16
#define LDS_BYTES ((size_t)131072)                     // 4 parts * 32 KB

// ---------------------------------------------------------------------------
// Pre-swizzle a 128x128 fp32 weight (row-major W[k][n]) into bf16 hi/lo WMMA
// B-fragments: element j of lane L of fragment (nt,kk) holds
// W[32*kk + Kmap(j,h)][nt*16 + (L&15)],  Kmap(j,h) = j + 8h + (j>=8 ? 8 : 0),
// matching CDNA5 16-bit B-matrix VGPR striping.
// ---------------------------------------------------------------------------
__global__ __launch_bounds__(32)
void sage_prep_w(const float* __restrict__ W, __bf16* __restrict__ whi,
                 __bf16* __restrict__ wlo) {
  const int lane = threadIdx.x;
  const int kk = blockIdx.x & 3;
  const int nt = blockIdx.x >> 2;
  const int n  = nt * 16 + (lane & 15);
  const int h  = lane >> 4;
  __bf16* ph = whi + (size_t)((nt * 4 + kk) * 32 + lane) * 16;
  __bf16* pl = wlo + (size_t)((nt * 4 + kk) * 32 + lane) * 16;
#pragma unroll
  for (int j = 0; j < 16; ++j) {
    const int K = 32 * kk + j + 8 * h + ((j >= 8) ? 8 : 0);
    const float f = W[(size_t)K * D + n];
    const __bf16 bh = (__bf16)f;            // RNE hi part
    const float rem = f - (float)bh;        // exact residual
    ph[j] = bh;
    pl[j] = (__bf16)rem;
  }
}

// ---------------------------------------------------------------------------
// Edge scatter: one wave32 per edge. Lane L moves features [4L, 4L+4) of
// x[src] into agg[dst] via global_atomic_add_f32 (L2-resident: 51MB < 192MB).
// Lane 0 optionally bumps deg[dst].
// ---------------------------------------------------------------------------
__global__ __launch_bounds__(256)
void sage_scatter(const float* __restrict__ feat, const int* __restrict__ src,
                  const int* __restrict__ dst, float* __restrict__ agg,
                  float* __restrict__ deg, int E) {
  const int gid  = blockIdx.x * blockDim.x + threadIdx.x;
  const int e    = gid >> 5;
  const int lane = gid & 31;
  if (e >= E) return;
  const int s = src[e];
  const int d = dst[e];
  const float4 v = *(const float4*)(feat + (size_t)s * D + lane * 4);
  float* base = agg + (size_t)d * D + lane * 4;
  atomicAdd(base + 0, v.x);
  atomicAdd(base + 1, v.y);
  atomicAdd(base + 2, v.z);
  atomicAdd(base + 3, v.w);
  if (deg != nullptr && lane == 0) atomicAdd(deg + d, 1.0f);
}

// ---------------------------------------------------------------------------
// Load one 16x32 bf16 A-fragment (hi + residual lo) from a row-major fp32
// matrix, applying `scale` (mean = agg * 1/deg).
// Lane layout: row M = lane&15, half h = lane>>4;
//   elements j=0..7  <- K = 32*kk + 8h     .. +7   (contiguous float8)
//   elements j=8..15 <- K = 32*kk + 16+8h  .. +7   (contiguous float8)
// ---------------------------------------------------------------------------
__device__ __forceinline__
void load_a_frag(const float* __restrict__ rowptr, int kk, int h, float scale,
                 v16bf& hi, v16bf& lo) {
  const float4 c0 = *(const float4*)(rowptr + kk * 32 + 8 * h);
  const float4 c1 = *(const float4*)(rowptr + kk * 32 + 8 * h + 4);
  const float4 c2 = *(const float4*)(rowptr + kk * 32 + 16 + 8 * h);
  const float4 c3 = *(const float4*)(rowptr + kk * 32 + 16 + 8 * h + 4);
  float f[16] = {c0.x, c0.y, c0.z, c0.w, c1.x, c1.y, c1.z, c1.w,
                 c2.x, c2.y, c2.z, c2.w, c3.x, c3.y, c3.z, c3.w};
#pragma unroll
  for (int j = 0; j < 16; ++j) {
    const float v = f[j] * scale;
    const __bf16 bh = (__bf16)v;
    hi[j] = bh;
    lo[j] = (__bf16)(v - (float)bh);
  }
}

// ---------------------------------------------------------------------------
// Fused SAGEConv: out = relu?( mean @ W_l + x @ W_r + b )
// 10 waves per block; the block first stages all 4 fragment buffers
// (Wl_hi, Wl_lo, Wr_hi, Wr_lo = 128 KB) into LDS once (2 blocks/WGP under the
// 320 KB WGP LDS), then each wave computes one 16-row x 128-col stripe with
// B-fragments served by ds_load_b128 (stride-32B per lane: bank-conflict-free).
// 6250 M-tiles / 10 waves = 625 blocks exactly -> EXEC all-ones for WMMA.
// bf16 hi/lo split -> 6 v_wmma_f32_16x16x32_bf16 per (tile, K-slice).
// ---------------------------------------------------------------------------
__global__ __launch_bounds__(320)
void sage_gemm(const float* __restrict__ xin, const float* __restrict__ agg,
               const float* __restrict__ deg,
               const __bf16* __restrict__ wl_hi, const __bf16* __restrict__ wl_lo,
               const __bf16* __restrict__ wr_hi, const __bf16* __restrict__ wr_lo,
               const float* __restrict__ bias, float* __restrict__ out,
               int do_relu) {
  extern __shared__ char smem_raw[];
  __bf16* smem = (__bf16*)smem_raw;

  // ---- Stage 128 KB of swizzled weight fragments into LDS (once/block) ----
  {
    uint4* dst = (uint4*)smem;                       // 8192 16B chunks total
    const __bf16* parts[1];
    (void)parts;
#pragma unroll
    for (int p = 0; p < 4; ++p) {
      const __bf16* sp = (p == 0) ? wl_hi : (p == 1) ? wl_lo
                        : (p == 2) ? wr_hi : wr_lo;
      const uint4* s4 = (const uint4*)sp;
      uint4* d4 = dst + (size_t)p * 2048;            // 32 KB per part
      for (int c = threadIdx.x; c < 2048; c += 320) d4[c] = s4[c];
    }
  }
  __syncthreads();

  const __bf16* s_wl_hi = smem;
  const __bf16* s_wl_lo = smem + 1 * WF_ELEMS;
  const __bf16* s_wr_hi = smem + 2 * WF_ELEMS;
  const __bf16* s_wr_lo = smem + 3 * WF_ELEMS;

  const int lane = threadIdx.x & 31;
  const int wave = threadIdx.x >> 5;                 // 0..9
  const int rloc = lane & 15;
  const int h    = lane >> 4;
  const int tile = blockIdx.x * 10 + wave;           // 0..6249, exact
  const int row  = tile * 16 + rloc;

  const float invd = 1.0f / fmaxf(deg[row], 1.0f);
  const float* aggrow = agg + (size_t)row * D;
  const float* xrow   = xin + (size_t)row * D;

  const v8f zero = {};
  v8f acc[8];
#pragma unroll
  for (int nt = 0; nt < 8; ++nt) acc[nt] = zero;

#pragma unroll
  for (int kk = 0; kk < 4; ++kk) {
    v16bf am_hi, am_lo, ax_hi, ax_lo;
    load_a_frag(aggrow, kk, h, invd, am_hi, am_lo);
    load_a_frag(xrow,   kk, h, 1.0f, ax_hi, ax_lo);
#pragma unroll
    for (int nt = 0; nt < 8; ++nt) {
      const size_t fo = (size_t)((nt * 4 + kk) * 32 + lane) * 16;
      const v16bf bl_hi = *(const v16bf*)(s_wl_hi + fo);
      const v16bf bl_lo = *(const v16bf*)(s_wl_lo + fo);
      const v16bf br_hi = *(const v16bf*)(s_wr_hi + fo);
      const v16bf br_lo = *(const v16bf*)(s_wr_lo + fo);
      acc[nt] = __builtin_amdgcn_wmma_f32_16x16x32_bf16(
          false, am_hi, false, bl_hi, (short)0, acc[nt], false, false);
      acc[nt] = __builtin_amdgcn_wmma_f32_16x16x32_bf16(
          false, am_lo, false, bl_hi, (short)0, acc[nt], false, false);
      acc[nt] = __builtin_amdgcn_wmma_f32_16x16x32_bf16(
          false, am_hi, false, bl_lo, (short)0, acc[nt], false, false);
      acc[nt] = __builtin_amdgcn_wmma_f32_16x16x32_bf16(
          false, ax_hi, false, br_hi, (short)0, acc[nt], false, false);
      acc[nt] = __builtin_amdgcn_wmma_f32_16x16x32_bf16(
          false, ax_lo, false, br_hi, (short)0, acc[nt], false, false);
      acc[nt] = __builtin_amdgcn_wmma_f32_16x16x32_bf16(
          false, ax_hi, false, br_lo, (short)0, acc[nt], false, false);
    }
  }

  // Epilogue: C/D layout -> element (M = v + 8h, N = nt*16 + rloc)
#pragma unroll
  for (int nt = 0; nt < 8; ++nt) {
    const int col = nt * 16 + rloc;
    const float bv = bias[col];
#pragma unroll
    for (int v = 0; v < 8; ++v) {
      float o = acc[nt][v] + bv;
      if (do_relu) o = fmaxf(o, 0.0f);
      out[(size_t)(tile * 16 + v + 8 * h) * D + col] = o;
    }
  }
}

// ---------------------------------------------------------------------------
// Launch: inputs (setup_inputs order):
//   0:x[N,128] f32  1:edge_index[2,E] int  2:W_l0  3:W_r0  4:b0
//   5:W_l1  6:W_r1  7:b1       out: [N,128] f32
// ---------------------------------------------------------------------------
extern "C" void kernel_launch(void* const* d_in, const int* in_sizes, int n_in,
                              void* d_out, int out_size, void* d_ws, size_t ws_size,
                              hipStream_t stream) {
  const float* x   = (const float*)d_in[0];
  const int*   ei  = (const int*)d_in[1];
  const float* Wl0 = (const float*)d_in[2];
  const float* Wr0 = (const float*)d_in[3];
  const float* b0  = (const float*)d_in[4];
  const float* Wl1 = (const float*)d_in[5];
  const float* Wr1 = (const float*)d_in[6];
  const float* b1  = (const float*)d_in[7];
  const int E = in_sizes[1] / 2;

  char* ws = (char*)d_ws;
  float*  agg  = (float*)(ws + OFF_AGG);
  float*  deg  = (float*)(ws + OFF_DEG);
  float*  hbuf = (float*)(ws + OFF_H);
  __bf16* wf   = (__bf16*)(ws + OFF_WF);
  __bf16* wl0h = wf + 0 * WF_ELEMS;  __bf16* wl0l = wf + 1 * WF_ELEMS;
  __bf16* wr0h = wf + 2 * WF_ELEMS;  __bf16* wr0l = wf + 3 * WF_ELEMS;
  __bf16* wl1h = wf + 4 * WF_ELEMS;  __bf16* wl1l = wf + 5 * WF_ELEMS;
  __bf16* wr1h = wf + 6 * WF_ELEMS;  __bf16* wr1l = wf + 7 * WF_ELEMS;

  const int scatter_blocks = (E + 7) / 8;   // 8 edges (waves) per 256-thread block
  const int gemm_blocks    = N_NODES / (16 * 10);  // 625, exact

  // Weight pre-swizzle (tiny, once per launch; deterministic)
  sage_prep_w<<<32, 32, 0, stream>>>(Wl0, wl0h, wl0l);
  sage_prep_w<<<32, 32, 0, stream>>>(Wr0, wr0h, wr0l);
  sage_prep_w<<<32, 32, 0, stream>>>(Wl1, wl1h, wl1l);
  sage_prep_w<<<32, 32, 0, stream>>>(Wr1, wr1h, wr1l);

  // ---- Layer 0 ----
  hipMemsetAsync(agg, 0, SZ_AGG, stream);
  hipMemsetAsync(deg, 0, (size_t)N_NODES * 4, stream);
  sage_scatter<<<scatter_blocks, 256, 0, stream>>>(x, ei, ei + E, agg, deg, E);
  sage_gemm<<<gemm_blocks, 320, LDS_BYTES, stream>>>(
      x, agg, deg, wl0h, wl0l, wr0h, wr0l, b0, hbuf, /*relu=*/1);

  // ---- Layer 1 (deg unchanged: same dst list) ----
  hipMemsetAsync(agg, 0, SZ_AGG, stream);
  sage_scatter<<<scatter_blocks, 256, 0, stream>>>(hbuf, ei, ei + E, agg, nullptr, E);
  sage_gemm<<<gemm_blocks, 320, LDS_BYTES, stream>>>(
      hbuf, agg, deg, wl1h, wl1l, wr1h, wr1l, b1, (float*)d_out, /*relu=*/0);
}
